// TokenPruningViT_31490700214572
// MI455X (gfx1250) — matrix operand order repaired and checked by reading, not compiled, not run on hardware
//
#include <hip/hip_runtime.h>
#include <hip/hip_bf16.h>
#include <math.h>

// ---------------- model constants ----------------
#define BD   32      // batch
#define LD   12      // layers
#define DM   768     // model dim
#define HH   12      // heads
#define HDM  64      // head dim
#define NTOK0 197    // tokens pre-prune
#define NPATCH 196
#define NRET 112     // retained tokens (8 pivots + 8*13)
#define NTOK1 113    // tokens post-prune (1 cls + 112)
#define PRUNE_L 6
#define ATT_SCALE 0.125f   // 64^-0.5
#define NPAD 224     // padded token count for attention tiles (14 * 16)
#define VSTR 232     // Vst row stride (tokens), 464B = 16B aligned

typedef __bf16 bf16;
typedef __attribute__((ext_vector_type(16))) __bf16 v16bf;
typedef __attribute__((ext_vector_type(8)))  float  v8f;
typedef __attribute__((ext_vector_type(4)))  unsigned u32x4;
typedef __attribute__((ext_vector_type(8)))  int      i32x8;
typedef __attribute__((ext_vector_type(4)))  int      i32x4;
typedef int gvi4 __attribute__((vector_size(16)));   // matches async-LDS builtin param

// -------- CDNA5 feature detection (compile-time, safe fallbacks) --------
#if __has_builtin(__builtin_amdgcn_global_load_async_to_lds_b128) && __has_builtin(__builtin_amdgcn_s_wait_asynccnt)
#define HAVE_ASYNC 1
#else
#define HAVE_ASYNC 0
#endif
#if __has_builtin(__builtin_amdgcn_tensor_load_to_lds) && __has_builtin(__builtin_amdgcn_s_wait_tensorcnt)
#define HAVE_TDM 1
#else
#define HAVE_TDM 0
#endif
#define ASY(p) ((gvi4*)(void*)(p))   // builtin takes generic int4* for src and dst

__device__ __forceinline__ float bf2f(bf16 v) {
  unsigned short s = __builtin_bit_cast(unsigned short, v);
  unsigned u = ((unsigned)s) << 16;
  return __builtin_bit_cast(float, u);
}
__device__ __forceinline__ bf16 f2bf(float f) {
  unsigned u = __builtin_bit_cast(unsigned, f);
  u += 0x7FFFu + ((u >> 16) & 1u);           // round-to-nearest-even
  unsigned short s = (unsigned short)(u >> 16);
  return __builtin_bit_cast(bf16, s);
}
// Build a 16-element bf16 fragment from two 16B LDS chunks (2x ds_load_b128).
__device__ __forceinline__ v16bf frag_ld(const bf16* p0, const bf16* p1) {
  union { uint4 u[2]; v16bf v; } t;
  t.u[0] = *reinterpret_cast<const uint4*>(p0);
  t.u[1] = *reinterpret_cast<const uint4*>(p1);
  return t.v;
}

#if HAVE_TDM
__device__ __forceinline__ unsigned lds_off(const void* p) {
  return (unsigned)(size_t)(__attribute__((address_space(3))) const void*)p;
}
#endif

// ---------------- fp32 -> bf16 transposed weight conversion -------------------
// dst[n*K + k] = bf16(src[k*N + n])  (weights stored as W^T, N x K, K contiguous)
__global__ void convt_kernel(const float* __restrict__ src, bf16* __restrict__ dst,
                             int K, int N) {
  size_t id = (size_t)blockIdx.x * blockDim.x + threadIdx.x;
  if (id >= (size_t)K * N) return;
  int n = (int)(id % N), k = (int)(id / N);
  dst[(size_t)n * K + k] = f2bf(src[id]);
}

// ---------------- im2col patch extraction ------------------------------------
__global__ void patch_kernel(const float* __restrict__ x, bf16* __restrict__ patches) {
  size_t id = (size_t)blockIdx.x * blockDim.x + threadIdx.x;
  const size_t total = (size_t)BD * NPATCH * DM;
  if (id >= total) return;
  int col = (int)(id % DM);
  int p   = (int)((id / DM) % NPATCH);
  int b   = (int)(id / ((size_t)DM * NPATCH));
  int c = col >> 8, rem = col & 255, i = rem >> 4, j = rem & 15;
  int gy = p / 14, gx = p % 14;
  float v = x[(((size_t)b * 3 + c) * 224 + (gy * 16 + i)) * 224 + gx * 16 + j];
  patches[id] = f2bf(v);
}

// ---------------- t = concat(cls, patch_embed) + pos_embed --------------------
__global__ void assemble_kernel(const float* __restrict__ pe, const float* __restrict__ cls,
                                const float* __restrict__ pos, float* __restrict__ t0) {
  size_t id = (size_t)blockIdx.x * blockDim.x + threadIdx.x;
  const size_t total = (size_t)BD * NTOK0 * DM;
  if (id >= total) return;
  int c   = (int)(id % DM);
  int tok = (int)((id / DM) % NTOK0);
  int b   = (int)(id / ((size_t)DM * NTOK0));
  float v = (tok == 0) ? cls[c] : pe[((size_t)b * NPATCH + (tok - 1)) * DM + c];
  t0[id] = v + pos[(size_t)tok * DM + c];
}

// ---------------- LayerNorm over D=768, out bf16 (mode 0) or f32 (mode 1) -----
__global__ __launch_bounds__(256)
void ln_kernel(const float* __restrict__ x, const float* __restrict__ g,
               const float* __restrict__ bta, bf16* __restrict__ outb,
               float* __restrict__ outf, int mode) {
  __shared__ float red[256];
  int row = blockIdx.x, tid = threadIdx.x;
  const float* xr = x + (size_t)row * DM;
  float s = 0.f;
  for (int d = tid; d < DM; d += 256) s += xr[d];
  red[tid] = s; __syncthreads();
  for (int o = 128; o > 0; o >>= 1) { if (tid < o) red[tid] += red[tid + o]; __syncthreads(); }
  float mean = red[0] * (1.0f / DM); __syncthreads();
  float v = 0.f;
  for (int d = tid; d < DM; d += 256) { float t = xr[d] - mean; v += t * t; }
  red[tid] = v; __syncthreads();
  for (int o = 128; o > 0; o >>= 1) { if (tid < o) red[tid] += red[tid + o]; __syncthreads(); }
  float rstd = rsqrtf(red[0] * (1.0f / DM) + 1e-6f);
  for (int d = tid; d < DM; d += 256) {
    float y = (xr[d] - mean) * rstd * g[d] + bta[d];
    if (mode == 0) outb[(size_t)row * DM + d] = f2bf(y);
    else           outf[(size_t)row * DM + d] = y;
  }
}

// ---------------- WMMA bf16 GEMM: C[MxN] = A[MxK] * (WT[NxK])^T + bias --------
// WT is the transposed weight (N x K, K contiguous). 64x128 block tile, 8 waves,
// 4 accumulators per wave. outMode: 0 f32 out, 1 bf16 out, 2 f32 residual +=.
__global__ __launch_bounds__(256)
void gemm_bf16(const bf16* __restrict__ A, const bf16* __restrict__ WT,
               const float* __restrict__ bias, float* __restrict__ Cf,
               bf16* __restrict__ Cb, int M, int N, int K, int outMode, int doGelu) {
  __shared__ bf16 As[64 * 40];    // 64 rows x 32 k (pad 8) : 80B rows
  __shared__ bf16 Bs[128 * 40];   // 128 cols x 32 k (pad 8)
  int tid = threadIdx.x, w = tid >> 5, lane = tid & 31;
  int bm = blockIdx.y * 64, bn = blockIdx.x * 128;
  int wr = (w & 3) * 16;          // wave row tile
  int wc = (w >> 2) * 16;         // wave col tiles: wc + {0,32,64,96}
  v8f z8 = {};
  v8f acc0 = z8, acc1 = z8, acc2 = z8, acc3 = z8;
  int arow = tid >> 2, aseg = (tid & 3) * 8;   // A loader: 64x32 / 256thr
  int r16 = lane & 15, nb = lane & 15;
  int ab = (lane < 16) ? 0 : 16;   // A fragment half select (contiguous k = 16h + s)
  int kb = (lane < 16) ? 0 : 16;   // B fragment K half select

  for (int k0 = 0; k0 < K; k0 += 32) {
    __syncthreads();
    // ---- stage A tile (row-major, K contiguous) ----
    int gr = bm + arow;
    const bf16* ap = A + (size_t)gr * K + k0 + aseg;
#if HAVE_ASYNC
    if (gr < M) {
      __builtin_amdgcn_global_load_async_to_lds_b128(ASY(ap), ASY(&As[arow * 40 + aseg]), 0, 0);
    } else {
      uint4 z = {0u,0u,0u,0u};
      *reinterpret_cast<uint4*>(&As[arow * 40 + aseg]) = z;
    }
#else
    uint4 av = {0u,0u,0u,0u};
    if (gr < M) {
      av = *reinterpret_cast<const uint4*>(ap);
      if (k0 + 32 < K) __builtin_prefetch(ap + 32, 0, 0);   // global_prefetch_b8
    }
    *reinterpret_cast<uint4*>(&As[arow * 40 + aseg]) = av;
#endif
    // ---- stage B tile: 128 rows (cols of C) x 32 k from WT ----
#if HAVE_TDM
    if (w == 0) {
      // Tensor Data Mover: 2D tile 32 elems x 128 rows, bf16, padded 32+8 per row
      unsigned long long ga = (unsigned long long)(size_t)(WT + (size_t)bn * K + k0);
      u32x4 g0; i32x8 g1; i32x4 gz; i32x8 gz8;
      gz[0] = gz[1] = gz[2] = gz[3] = 0;
      gz8[0] = gz8[1] = gz8[2] = gz8[3] = gz8[4] = gz8[5] = gz8[6] = gz8[7] = 0;
      g0[0] = 1u;                                    // count=1 valid descriptor
      g0[1] = lds_off(Bs);                           // lds_addr
      g0[2] = (unsigned)ga;                          // global_addr[31:0]
      g0[3] = (unsigned)((ga >> 32) & 0x1FFFFFFu) | (2u << 30);  // addr[56:32] | type=2
      g1[0] = (1 << 16)    // data_size = 2B
            | (1 << 20)    // pad_enable
            | (3 << 22)    // pad_interval: 16 DWORDs (=32 bf16)
            | (3 << 25);   // pad_amount:   4 DWORDs (=8 bf16)
      g1[1] = (int)(((unsigned)K & 0xFFFFu) << 16);                 // tensor_dim0[15:0]
      g1[2] = (int)((((unsigned)K >> 16) & 0xFFFFu)                 // tensor_dim0[31:16]
            | (128u << 16));                                        // tensor_dim1[15:0]
      g1[3] = (int)(32u << 16);                                     // tile_dim0 = 32
      g1[4] = 128;                                                  // tile_dim1 = 128
      g1[5] = (int)K;                                               // tensor_dim0_stride
      g1[6] = 0; g1[7] = 0;
      __builtin_amdgcn_tensor_load_to_lds(g0, g1, gz, gz, gz8, 0);  // 6-arg form
      __builtin_amdgcn_s_wait_tensorcnt(0);
    }
#else
    {
      int brow = tid >> 2, kseg = (tid & 3) * 8;     // 2 rows per thread
      #pragma unroll
      for (int rr = 0; rr < 2; ++rr) {
        int rn = brow + rr * 64;
        const bf16* bp = WT + (size_t)(bn + rn) * K + k0 + kseg;
#if HAVE_ASYNC
        __builtin_amdgcn_global_load_async_to_lds_b128(ASY(bp), ASY(&Bs[rn * 40 + kseg]), 0, 0);
#else
        *reinterpret_cast<uint4*>(&Bs[rn * 40 + kseg]) = *reinterpret_cast<const uint4*>(bp);
#endif
      }
    }
#endif
#if HAVE_ASYNC
    __builtin_amdgcn_s_wait_asynccnt(0);
#endif
    __syncthreads();
    // ---- fragments: 2x ds_load_b128 each ----
    const bf16* arowp = &As[(wr + r16) * 40];
    v16bf af = frag_ld(arowp + ab, arowp + ab + 8);
    v16bf b0 = frag_ld(&Bs[(wc +  0 + nb) * 40 + kb], &Bs[(wc +  0 + nb) * 40 + kb + 8]);
    v16bf b1 = frag_ld(&Bs[(wc + 32 + nb) * 40 + kb], &Bs[(wc + 32 + nb) * 40 + kb + 8]);
    v16bf b2 = frag_ld(&Bs[(wc + 64 + nb) * 40 + kb], &Bs[(wc + 64 + nb) * 40 + kb + 8]);
    v16bf b3 = frag_ld(&Bs[(wc + 96 + nb) * 40 + kb], &Bs[(wc + 96 + nb) * 40 + kb + 8]);
    acc0 = __builtin_amdgcn_wmma_f32_16x16x32_bf16(false, af, false, b0, (short)0, acc0, false, false);
    acc1 = __builtin_amdgcn_wmma_f32_16x16x32_bf16(false, af, false, b1, (short)0, acc1, false, false);
    acc2 = __builtin_amdgcn_wmma_f32_16x16x32_bf16(false, af, false, b2, (short)0, acc2, false, false);
    acc3 = __builtin_amdgcn_wmma_f32_16x16x32_bf16(false, af, false, b3, (short)0, acc3, false, false);
  }
  int rb = (lane < 16) ? 0 : 8;
  #pragma unroll
  for (int t = 0; t < 4; ++t) {
    v8f a = (t == 0) ? acc0 : (t == 1) ? acc1 : (t == 2) ? acc2 : acc3;
    int gn = bn + wc + t * 32 + nb;
    float bv = bias[gn];
    #pragma unroll
    for (int r = 0; r < 8; ++r) {
      int gm = bm + wr + rb + r;
      if (gm >= M) continue;
      float v = a[r] + bv;
      if (doGelu) v = 0.5f * v * (1.0f + erff(v * 0.70710678118f));
      size_t idx = (size_t)gm * N + gn;
      if (outMode == 0)      Cf[idx] = v;
      else if (outMode == 1) Cb[idx] = f2bf(v);
      else                   Cf[idx] += v;
    }
  }
}

// ---------------- fused attention per (batch, head), all-WMMA -----------------
__global__ __launch_bounds__(256)
void attn_kernel(const bf16* __restrict__ qkv, bf16* __restrict__ attn_out, int Ntok) {
  extern __shared__ __align__(16) char smem[];
  bf16* Qs  = (bf16*)smem;                 // [NPAD][72] row-major per token
  bf16* Ks  = Qs + NPAD * 72;              // [NPAD][72]
  bf16* Vst = Ks + NPAD * 72;              // [64 cols][VSTR toks] (transposed)
  float* Sb = (float*)(Vst + 64 * VSTR);   // per-wave 16 x NPAD f32 scores
  bf16* Pb  = (bf16*)(Sb + 8 * 16 * NPAD); // per-wave 16 x NPAD bf16 probs
  int tid = threadIdx.x, w = tid >> 5, lane = tid & 31;
  int b = blockIdx.x / HH, h = blockIdx.x % HH;
  size_t base = (size_t)b * Ntok * (3 * DM) + h * HDM;

  for (int idx = tid; idx < NPAD * 8; idx += 256) {   // stage Q/K/V head slices
    int tok = idx >> 3, seg = (idx & 7) * 8;
    if (tok < Ntok) {
      size_t r = base + (size_t)tok * (3 * DM) + seg;
#if HAVE_ASYNC
      __builtin_amdgcn_global_load_async_to_lds_b128(ASY(qkv + r),      ASY(&Qs[tok * 72 + seg]), 0, 0);
      __builtin_amdgcn_global_load_async_to_lds_b128(ASY(qkv + r + DM), ASY(&Ks[tok * 72 + seg]), 0, 0);
#else
      *reinterpret_cast<uint4*>(&Qs[tok * 72 + seg]) = *reinterpret_cast<const uint4*>(qkv + r);
      *reinterpret_cast<uint4*>(&Ks[tok * 72 + seg]) = *reinterpret_cast<const uint4*>(qkv + r + DM);
#endif
      union { uint4 u; bf16 hlf[8]; } vv;
      vv.u = *reinterpret_cast<const uint4*>(qkv + r + 2 * DM);
      #pragma unroll
      for (int e = 0; e < 8; ++e) Vst[(seg + e) * VSTR + tok] = vv.hlf[e];   // transpose V
    } else {
      uint4 z = {0u,0u,0u,0u};
      *reinterpret_cast<uint4*>(&Qs[tok * 72 + seg]) = z;
      *reinterpret_cast<uint4*>(&Ks[tok * 72 + seg]) = z;
      bf16 zb = f2bf(0.f);
      #pragma unroll
      for (int e = 0; e < 8; ++e) Vst[(seg + e) * VSTR + tok] = zb;
    }
  }
#if HAVE_ASYNC
  __builtin_amdgcn_s_wait_asynccnt(0);
#endif
  __syncthreads();

  float* Sw = Sb + w * 16 * NPAD;
  bf16*  Pw = Pb + w * 16 * NPAD;
  int r16 = lane & 15, nb = lane & 15;
  int ab = (lane < 16) ? 0 : 16, kb = (lane < 16) ? 0 : 16, rb = (lane < 16) ? 0 : 8;

  for (int qt = w; qt < NPAD / 16; qt += 8) {
    if (qt * 16 >= Ntok) continue;                     // wave-uniform skip
    // ---- S = (Q K^T) * scale ----
    for (int jt = 0; jt < NPAD / 16; ++jt) {
      v8f accS = {};
      #pragma unroll
      for (int kc = 0; kc < HDM; kc += 32) {
        const bf16* qp = &Qs[(qt * 16 + r16) * 72 + kc];
        v16bf af = frag_ld(qp + ab, qp + ab + 8);
        const bf16* kp = &Ks[(jt * 16 + nb) * 72 + kc + kb];
        v16bf bf = frag_ld(kp, kp + 8);
        accS = __builtin_amdgcn_wmma_f32_16x16x32_bf16(false, af, false, bf, (short)0, accS, false, false);
      }
      #pragma unroll
      for (int r = 0; r < 8; ++r) Sw[(rb + r) * NPAD + jt * 16 + nb] = accS[r] * ATT_SCALE;
    }
    // ---- per-row softmax: S(f32) -> P(bf16) ----
    if (lane < 16) {
      float* Srow = Sw + lane * NPAD;
      bf16*  Prow = Pw + lane * NPAD;
      int qrow = qt * 16 + lane;
      if (qrow < Ntok) {
        float mx = -1e30f;
        for (int c = 0; c < Ntok; ++c) mx = fmaxf(mx, Srow[c]);
        float sm = 0.f;
        for (int c = 0; c < Ntok; ++c) { float e = __expf(Srow[c] - mx); Srow[c] = e; sm += e; }
        float inv = 1.0f / sm;
        for (int c = 0; c < Ntok; ++c) Prow[c] = f2bf(Srow[c] * inv);
      } else {
        for (int c = 0; c < Ntok; ++c) Prow[c] = f2bf(0.f);
      }
      for (int c = Ntok; c < NPAD; ++c) Prow[c] = f2bf(0.f);
    }
    // ---- O = P V ----
    v8f z8 = {};
    v8f o0 = z8, o1 = z8, o2 = z8, o3 = z8;
    for (int kc = 0; kc < NPAD; kc += 32) {
      const bf16* pp = &Pw[r16 * NPAD + kc];
      v16bf af = frag_ld(pp + ab, pp + ab + 8);
      const bf16* v0 = &Vst[( 0 + nb) * VSTR + kc + kb];
      const bf16* v1 = &Vst[(16 + nb) * VSTR + kc + kb];
      const bf16* v2 = &Vst[(32 + nb) * VSTR + kc + kb];
      const bf16* v3 = &Vst[(48 + nb) * VSTR + kc + kb];
      v16bf bf0 = frag_ld(v0, v0 + 8);
      v16bf bf1 = frag_ld(v1, v1 + 8);
      v16bf bf2 = frag_ld(v2, v2 + 8);
      v16bf bf3 = frag_ld(v3, v3 + 8);
      o0 = __builtin_amdgcn_wmma_f32_16x16x32_bf16(false, af, false, bf0, (short)0, o0, false, false);
      o1 = __builtin_amdgcn_wmma_f32_16x16x32_bf16(false, af, false, bf1, (short)0, o1, false, false);
      o2 = __builtin_amdgcn_wmma_f32_16x16x32_bf16(false, af, false, bf2, (short)0, o2, false, false);
      o3 = __builtin_amdgcn_wmma_f32_16x16x32_bf16(false, af, false, bf3, (short)0, o3, false, false);
    }
    #pragma unroll
    for (int ct = 0; ct < 4; ++ct) {
      v8f a = (ct == 0) ? o0 : (ct == 1) ? o1 : (ct == 2) ? o2 : o3;
      #pragma unroll
      for (int r = 0; r < 8; ++r) {
        int tok = qt * 16 + rb + r;
        if (tok < Ntok)
          attn_out[((size_t)b * Ntok + tok) * DM + h * HDM + ct * 16 + nb] = f2bf(a[r]);
      }
    }
  }
}

// ---------------- DART token selection (one block per sample) -----------------
__global__ __launch_bounds__(256)
void dart_kernel(const bf16* __restrict__ qkv, const float* __restrict__ last,
                 int* __restrict__ retained) {
  __shared__ float l1[NTOK0], inv[NTOK0], cs[NTOK0];
  __shared__ int piv[8];
  __shared__ unsigned char valid[NTOK0], sel[NTOK0];
  int b = blockIdx.x, tid = threadIdx.x;
  for (int tok = tid; tok < NTOK0; tok += 256) {
    const bf16* kp = qkv + ((size_t)b * NTOK0 + tok) * (3 * DM) + DM;
    float s = 0.f;
    for (int d = 0; d < DM; ++d) s += fabsf(bf2f(kp[d]));
    l1[tok] = s;
    const float* lp = last + ((size_t)b * NTOK0 + tok) * DM;
    float n2 = 0.f;
    for (int d = 0; d < DM; ++d) { float v = lp[d]; n2 += v * v; }
    inv[tok] = 1.0f / fmaxf(sqrtf(n2), 1e-8f);
    valid[tok] = (tok != 0); sel[tok] = 0;
  }
  __syncthreads();
  if (tid == 0) {                                   // 8 pivots = top-8 by |k|_1
    for (int j = 0; j < 8; ++j) {
      int best = 1; float bv = -1e30f;
      for (int t = 1; t < NTOK0; ++t) if (l1[t] > bv) { bv = l1[t]; best = t; }
      l1[best] = -1e30f; piv[j] = best; valid[best] = 0; sel[best] = 1;
    }
  }
  __syncthreads();
  for (int j = 0; j < 8; ++j) {
    int pv = piv[j];
    const float* pp = last + ((size_t)b * NTOK0 + pv) * DM;
    for (int tok = tid; tok < NTOK0; tok += 256) {
      const float* lp = last + ((size_t)b * NTOK0 + tok) * DM;
      float d = 0.f;
      for (int k = 0; k < DM; ++k) d += lp[k] * pp[k];
      cs[tok] = d * inv[tok] * inv[pv];
    }
    __syncthreads();
    if (tid == 0) {                                  // top-13 least-similar valid
      for (int s = 0; s < 13; ++s) {
        int best = -1; float bv = 1e30f;
        for (int t = 1; t < NTOK0; ++t) if (valid[t] && cs[t] < bv) { bv = cs[t]; best = t; }
        if (best >= 0) { sel[best] = 1; valid[best] = 0; }
      }
    }
    __syncthreads();
  }
  if (tid == 0) {
    int cnt = 0;
    for (int t = 1; t < NTOK0 && cnt < NRET; ++t) if (sel[t]) retained[b * NRET + cnt++] = t;
    for (; cnt < NRET; ++cnt) retained[b * NRET + cnt] = NPATCH;
  }
}

// ---------------- gather pruned tokens ----------------------------------------
__global__ void gather_kernel(const float* __restrict__ t0, const int* __restrict__ retained,
                              float* __restrict__ t1) {
  size_t id = (size_t)blockIdx.x * blockDim.x + threadIdx.x;
  const size_t total = (size_t)BD * NTOK1 * DM;
  if (id >= total) return;
  int c = (int)(id % DM);
  int j = (int)((id / DM) % NTOK1);
  int b = (int)(id / ((size_t)DM * NTOK1));
  int src = (j == 0) ? 0 : retained[b * NRET + j - 1];
  t1[id] = t0[((size_t)b * NTOK0 + src) * DM + c];
}

// ---------------- classifier head ---------------------------------------------
__global__ void head_kernel(const float* __restrict__ lastf, const float* __restrict__ head_w,
                            const float* __restrict__ head_b, float* __restrict__ out, int Ntok) {
  int b = blockIdx.x, c = threadIdx.x;
  if (c >= 10) return;
  const float* row = lastf + (size_t)b * Ntok * DM;   // cls token row
  float s = head_b[c];
  for (int d = 0; d < DM; ++d) s += row[d] * head_w[d * 10 + c];
  out[b * 10 + c] = s;
}

// =============================== host side ====================================
extern "C" void kernel_launch(void* const* d_in, const int* in_sizes, int n_in,
                              void* d_out, int out_size, void* d_ws, size_t ws_size,
                              hipStream_t stream) {
  (void)in_sizes; (void)n_in; (void)out_size; (void)ws_size;
  const float* x       = (const float*)d_in[0];
  const float* patch_w = (const float*)d_in[1];
  const float* patch_b = (const float*)d_in[2];
  const float* cls_tok = (const float*)d_in[3];
  const float* pos_emb = (const float*)d_in[4];
  const float* ln1_g   = (const float*)d_in[5];
  const float* ln1_b   = (const float*)d_in[6];
  const float* qkv_w   = (const float*)d_in[7];
  const float* qkv_b   = (const float*)d_in[8];
  const float* proj_w  = (const float*)d_in[9];
  const float* proj_b  = (const float*)d_in[10];
  const float* ln2_g   = (const float*)d_in[11];
  const float* ln2_b   = (const float*)d_in[12];
  const float* fc1_w   = (const float*)d_in[13];
  const float* fc1_b   = (const float*)d_in[14];
  const float* fc2_w   = (const float*)d_in[15];
  const float* fc2_b   = (const float*)d_in[16];
  const float* norm_g  = (const float*)d_in[17];
  const float* norm_b  = (const float*)d_in[18];
  const float* head_w  = (const float*)d_in[19];
  const float* head_b  = (const float*)d_in[20];

  size_t off = 0;
  auto carve = [&](size_t bytes) -> void* {
    off = (off + 255) & ~(size_t)255;
    void* p = (char*)d_ws + off;
    off += bytes;
    return p;
  };
  const size_t M1 = (size_t)BD * NTOK0;          // 6304
  float* t0      = (float*)carve(M1 * DM * 4);
  float* t1      = (float*)carve(M1 * DM * 4);
  float* pe      = (float*)carve((size_t)BD * NPATCH * DM * 4);
  float* lastb   = (float*)carve(M1 * DM * 4);
  bf16*  patches = (bf16*) carve((size_t)BD * NPATCH * DM * 2);
  bf16*  lnout   = (bf16*) carve(M1 * DM * 2);
  bf16*  qkvb    = (bf16*) carve(M1 * 3 * DM * 2);
  bf16*  attno   = (bf16*) carve(M1 * DM * 2);
  bf16*  fc1o    = (bf16*) carve(M1 * 4 * DM * 2);
  bf16*  wbuf    = (bf16*) carve((size_t)DM * 4 * DM * 2);
  int*   retained= (int*)  carve((size_t)BD * NRET * 4);

  auto convt = [&](const float* src, int K, int N) {
    size_t n = (size_t)K * N;
    convt_kernel<<<(unsigned)((n + 255) / 256), 256, 0, stream>>>(src, wbuf, K, N);
  };
  auto gemm = [&](const bf16* A, const float* bias, float* Cf, bf16* Cb,
                  int M, int N, int K, int mode, int gelu) {
    dim3 g(N / 128, (M + 63) / 64);
    gemm_bf16<<<g, 256, 0, stream>>>(A, wbuf, bias, Cf, Cb, M, N, K, mode, gelu);
  };

  // ---- patch embedding ----
  {
    size_t n = (size_t)BD * NPATCH * DM;
    patch_kernel<<<(unsigned)((n + 255) / 256), 256, 0, stream>>>(x, patches);
    convt(patch_w, DM, DM);
    gemm(patches, patch_b, pe, nullptr, BD * NPATCH, DM, DM, 0, 0);
    size_t n2 = (size_t)BD * NTOK0 * DM;
    assemble_kernel<<<(unsigned)((n2 + 255) / 256), 256, 0, stream>>>(pe, cls_tok, pos_emb, t0);
  }

  constexpr size_t ATTN_SMEM = (size_t)2 * NPAD * 72 * 2      // Qs, Ks
                             + (size_t)64 * VSTR * 2          // Vst
                             + (size_t)8 * 16 * NPAD * 4      // S (f32)
                             + (size_t)8 * 16 * NPAD * 2;     // P (bf16)

  float* t = t0;
  int Ntok = NTOK0, M = BD * NTOK0;
  for (int l = 0; l < LD; ++l) {
    // attention block
    ln_kernel<<<M, 256, 0, stream>>>(t, ln1_g + l * DM, ln1_b + l * DM, lnout, nullptr, 0);
    convt(qkv_w + (size_t)l * DM * 3 * DM, DM, 3 * DM);
    gemm(lnout, qkv_b + l * 3 * DM, nullptr, qkvb, M, 3 * DM, DM, 1, 0);
    attn_kernel<<<BD * HH, 256, ATTN_SMEM, stream>>>(qkvb, attno, Ntok);
    convt(proj_w + (size_t)l * DM * DM, DM, DM);
    gemm(attno, proj_b + l * DM, t, nullptr, M, DM, DM, 2, 0);   // residual add
    // MLP block
    ln_kernel<<<M, 256, 0, stream>>>(t, ln2_g + l * DM, ln2_b + l * DM, lnout, nullptr, 0);
    convt(fc1_w + (size_t)l * DM * 4 * DM, DM, 4 * DM);
    gemm(lnout, fc1_b + l * 4 * DM, nullptr, fc1o, M, 4 * DM, DM, 1, 1);  // +GELU
    convt(fc2_w + (size_t)l * 4 * DM * DM, 4 * DM, DM);
    gemm(fc1o, fc2_b + l * DM, t, nullptr, M, DM, 4 * DM, 2, 0);  // residual add
    // DART pruning after layer 6
    if (l == PRUNE_L) {
      ln_kernel<<<M, 256, 0, stream>>>(t, norm_g, norm_b, nullptr, lastb, 1);
      dart_kernel<<<BD, 256, 0, stream>>>(qkvb, lastb, retained);
      size_t n = (size_t)BD * NTOK1 * DM;
      gather_kernel<<<(unsigned)((n + 255) / 256), 256, 0, stream>>>(t, retained, t1);
      t = t1; Ntok = NTOK1; M = BD * NTOK1;
    }
  }
  // final LN + head
  ln_kernel<<<M, 256, 0, stream>>>(t, norm_g, norm_b, nullptr, lastb, 1);
  head_kernel<<<BD, 32, 0, stream>>>(lastb, head_w, head_b, (float*)d_out, Ntok);
}
